// Center2D_58162447122627
// MI455X (gfx1250) — matrix operand order repaired and checked by reading, not compiled, not run on hardware
//
#include <hip/hip_runtime.h>
#include <hip/hip_bf16.h>

typedef __attribute__((ext_vector_type(16))) _Float16 v16h;
typedef __attribute__((ext_vector_type(8)))  float    v8f;

// db2 lowpass (REC_LO); DEC application reduces to REC_LO in forward order.
__device__ __constant__ float c_g[4] = {
    0.48296291314469025f, 0.8365163037378079f,
    0.22414386804185735f, -0.12940952255092145f };

// ---------------------------------------------------------------------------
// Weight pre-pack: f32 (CO,CIN,3,3) -> f16 WMMA A-fragment order.
// Packed index: ((((coTile*9 + tap)*(CIN/32) + chunk)*32 + lane)*16 + e
//   co  = coTile*16 + (lane&15)
//   cin = chunk*32 + ((lane>>4)<<4) + e   [K-layout of A]
// Each lane's 16 halves are contiguous (32B) -> vector loads in the conv.
// ---------------------------------------------------------------------------
template<int CIN, int CO>
__global__ __launch_bounds__(256)
void pack_conv_w_kernel(const float* __restrict__ wgt, _Float16* __restrict__ out)
{
    const int total = CO * CIN * 9;
    int idx = blockIdx.x * 256 + threadIdx.x;
    if (idx >= total) return;
    const int NCH  = CIN / 32;
    int e      = idx & 15;
    int lane   = (idx >> 4) & 31;
    int rest   = idx >> 9;            // (coTile*9 + tap)*NCH + chunk
    int chunk  = rest % NCH;
    int rest2  = rest / NCH;
    int tap    = rest2 % 9;
    int coTile = rest2 / 9;
    int co  = coTile * 16 + (lane & 15);
    int cin = chunk * 32 + ((lane >> 4) << 4) + e;
    out[idx] = (_Float16)wgt[(co * CIN + cin) * 9 + tap];
}

// Deconv weights (128,64,2,2): GEMM row m = o*4+k*2+l so dw[cin*256 + m].
__global__ __launch_bounds__(256)
void pack_deconv_w_kernel(const float* __restrict__ dw, _Float16* __restrict__ out)
{
    const int total = 256 * 128;
    int idx = blockIdx.x * 256 + threadIdx.x;
    if (idx >= total) return;
    int e     = idx & 15;
    int lane  = (idx >> 4) & 31;
    int rest  = idx >> 9;             // mTile*4 + chunk  (K chunks = 128/32)
    int chunk = rest & 3;
    int mTile = rest >> 2;
    int m   = mTile * 16 + (lane & 15);
    int cin = chunk * 32 + ((lane >> 4) << 4) + e;
    out[idx] = (_Float16)dw[cin * 256 + m];
}

// ---------------------------------------------------------------------------
// Kernel 1: fused DWT2-LL (symmetric pad) + 2x2 maxpool.
// x: (N,C,256,256) f32 -> out: f16 NHWC (N,64,64,C)  [feeds conv1 B loads]
// ---------------------------------------------------------------------------
__global__ __launch_bounds__(256)
void dwt_pool_kernel(const float* __restrict__ x, _Float16* __restrict__ out,
                     int C, long total)
{
    long idx = (long)blockIdx.x * blockDim.x + threadIdx.x;
    if (idx >= total) return;
    int pw = (int)(idx & 63);
    int ph = (int)((idx >> 6) & 63);
    long nc = idx >> 12;                 // n*C + c
    int c = (int)(nc % C);
    int n = (int)(nc / C);
    const float* xin = x + nc * 65536;

    float best = -3.4e38f;
    #pragma unroll
    for (int dh = 0; dh < 2; ++dh) {
        int h = 2 * ph + dh;
        #pragma unroll
        for (int dw = 0; dw < 2; ++dw) {
            int w = 2 * pw + dw;
            float s = 0.f;
            #pragma unroll
            for (int jh = 0; jh < 4; ++jh) {
                int ih = 2 * h + jh - 2;
                if (ih < 0) ih = -ih - 1; else if (ih >= 256) ih = 511 - ih;
                const float* row = xin + ih * 256;
                float rs = 0.f;
                #pragma unroll
                for (int jw = 0; jw < 4; ++jw) {
                    int iw = 2 * w + jw - 2;
                    if (iw < 0) iw = -iw - 1; else if (iw >= 256) iw = 511 - iw;
                    rs = fmaf(c_g[jw], row[iw], rs);
                }
                s = fmaf(c_g[jh], rs, s);
            }
            best = fmaxf(best, s);
        }
    }
    out[(((long)(n * 64 + ph) * 64) + pw) * C + c] = (_Float16)best;
}

// ---------------------------------------------------------------------------
// 3x3 SAME conv as implicit GEMM on WMMA f32<=f16*f16.
// One wave = one 16(Cout) x 16(N*H*W) tile; all operand loads are 32B vectors.
// act: f16 NHWC (N,H,W,CIN)   apack: packed f16 A-fragments
// out: f32 NCHW (N,CO,H,W) (+bias)
// ---------------------------------------------------------------------------
template<int CIN, int CO>
__global__ __launch_bounds__(32)
void conv3_wmma_kernel(const _Float16* __restrict__ act,
                       const _Float16* __restrict__ apack,
                       const float* __restrict__ bias,
                       float* __restrict__ out,
                       int H, int W)
{
    const int NCH = CIN / 32;
    const int lane = threadIdx.x;
    const int half = lane >> 4;
    const int l16  = lane & 15;
    const int posBase = blockIdx.x * 16;
    const int coTile  = blockIdx.y;
    const int HW = H * W;

    const int p     = posBase + l16;
    const int n_img = p / HW;
    const int hw    = p - n_img * HW;
    const int oh    = hw / W;
    const int ow    = hw - oh * W;

    v8f acc = {};
    for (int tap = 0; tap < 9; ++tap) {
        const int kh = tap / 3, kw = tap - kh * 3;
        const int ih = oh + kh - 1;
        const int iw = ow + kw - 1;
        const bool valid = (ih >= 0) & (ih < H) & (iw >= 0) & (iw < W);
        const int ihc = min(max(ih, 0), H - 1);
        const int iwc = min(max(iw, 0), W - 1);
        const _Float16* bbase =
            act + (((long)(n_img * H + ihc) * W + iwc) * CIN + (half << 4));
        const _Float16* abase =
            apack + ((long)(coTile * 9 + tap) * NCH) * 512 + lane * 16;
        if (valid && tap < 8) {
            // prefetch next tap's activation row (lowers to global_prefetch_b8)
            __builtin_prefetch(bbase + (long)CIN, 0, 0);
        }
        #pragma unroll
        for (int chunk = 0; chunk < NCH; ++chunk) {
            v16h a = *(const v16h*)(abase + chunk * 512);
            v16h b = {};
            if (valid) b = *(const v16h*)(bbase + chunk * 32);
            acc = __builtin_amdgcn_wmma_f32_16x16x32_f16(
                      false, a, false, b, (short)0, acc, false, false);
        }
    }
    // D fragment: element r -> M = half*8 + r, N = l16
    #pragma unroll
    for (int r = 0; r < 8; ++r) {
        int co = coTile * 16 + (half << 3) + r;
        out[(long)(n_img * CO + co) * HW + (long)oh * W + ow] = acc[r] + bias[co];
    }
}

// ---------------------------------------------------------------------------
// Per-channel batchnorm statistics (mean, biased var). One block per channel.
// ---------------------------------------------------------------------------
__global__ __launch_bounds__(256)
void bn_stats_kernel(const float* __restrict__ x, float* __restrict__ mean,
                     float* __restrict__ var, int Nimg, int C, int HW)
{
    const int c = blockIdx.x;
    const long per = (long)Nimg * HW;
    float s = 0.f, s2 = 0.f;
    for (long i = threadIdx.x; i < per; i += blockDim.x) {
        long n = i / HW;
        long r = i - n * HW;
        float v = x[(n * C + c) * HW + r];
        s  += v;
        s2 = fmaf(v, v, s2);
    }
    __shared__ float sh[256], sh2[256];
    sh[threadIdx.x] = s; sh2[threadIdx.x] = s2;
    __syncthreads();
    for (int off = 128; off > 0; off >>= 1) {
        if (threadIdx.x < off) {
            sh[threadIdx.x]  += sh[threadIdx.x + off];
            sh2[threadIdx.x] += sh2[threadIdx.x + off];
        }
        __syncthreads();
    }
    if (threadIdx.x == 0) {
        float m = sh[0] / (float)per;
        mean[c] = m;
        var[c]  = sh2[0] / (float)per - m * m;
    }
}

// ---------------------------------------------------------------------------
// BN normalize + ReLU: read f32 NCHW conv output, write f16 NHWC activations.
// ---------------------------------------------------------------------------
__global__ __launch_bounds__(256)
void bn_relu_kernel(const float* __restrict__ x, _Float16* __restrict__ out16,
                    const float* __restrict__ mean, const float* __restrict__ var,
                    const float* __restrict__ g, const float* __restrict__ b,
                    int C, int H, int W, long total)
{
    long i = (long)blockIdx.x * blockDim.x + threadIdx.x;
    if (i >= total) return;
    const int HW = H * W;
    int hw = (int)(i % HW);
    int c  = (int)((i / HW) % C);
    int n  = (int)(i / ((long)C * HW));
    int h = hw / W, w = hw - h * W;
    float inv = rsqrtf(var[c] + 1e-5f);
    float y = (x[i] - mean[c]) * inv * g[c] + b[c];
    y = fmaxf(y, 0.f);
    out16[(((long)(n * H + h) * W) + w) * C + c] = (_Float16)y;
}

// ---------------------------------------------------------------------------
// 2x2 deconv upsample as 256 x (N*h*w) x 128 GEMM on WMMA.
// act: f16 NHWC (N,64,64,128)  dpack: packed A  out: f32 NCHW (N,64,128,128)
// ---------------------------------------------------------------------------
__global__ __launch_bounds__(32)
void deconv2_wmma_kernel(const _Float16* __restrict__ act,
                         const _Float16* __restrict__ dpack,
                         const float* __restrict__ db,
                         float* __restrict__ out)
{
    const int CIN = 128, HW = 4096;
    const int lane = threadIdx.x;
    const int half = lane >> 4;
    const int l16  = lane & 15;
    const int posBase = blockIdx.x * 16;
    const int mTile   = blockIdx.y;

    const int p     = posBase + l16;
    const int n_img = p >> 12;
    const int hw    = p & (HW - 1);
    const int oh    = hw >> 6;
    const int ow    = hw & 63;

    const _Float16* bbase = act + ((long)p * CIN + (half << 4));
    const _Float16* abase = dpack + (long)mTile * 4 * 512 + lane * 16;

    v8f acc = {};
    #pragma unroll
    for (int chunk = 0; chunk < 4; ++chunk) {
        v16h a = *(const v16h*)(abase + chunk * 512);
        v16h b = *(const v16h*)(bbase + chunk * 32);
        acc = __builtin_amdgcn_wmma_f32_16x16x32_f16(
                  false, a, false, b, (short)0, acc, false, false);
    }
    #pragma unroll
    for (int r = 0; r < 8; ++r) {
        int m = mTile * 16 + (half << 3) + r;
        int o = m >> 2, k = (m >> 1) & 1, l = m & 1;
        out[((long)(n_img * 64 + o) * 128 + (2 * oh + k)) * 128 + (2 * ow + l)]
            = acc[r] + db[o];
    }
}

// ---------------------------------------------------------------------------
// Inverse DWT2-LL. a: (NC,128,128) f32 -> out: (NC,254,254) f32
// ---------------------------------------------------------------------------
__global__ __launch_bounds__(256)
void idwt2_kernel(const float* __restrict__ a, float* __restrict__ out, long total)
{
    long idx = (long)blockIdx.x * blockDim.x + threadIdx.x;
    if (idx >= total) return;
    int tw = (int)(idx % 254);
    long t1 = idx / 254;
    int th = (int)(t1 % 254);
    long nc = t1 / 254;
    const float* ain = a + nc * (128 * 128);

    float acc = 0.f;
    #pragma unroll
    for (int jh = 0; jh < 4; ++jh) {
        int qh = th + jh - 1;
        if (qh < 0 || qh > 254 || (qh & 1)) continue;
        const float* row = ain + (qh >> 1) * 128;
        float ch = c_g[3 - jh];
        #pragma unroll
        for (int jw = 0; jw < 4; ++jw) {
            int qw = tw + jw - 1;
            if (qw < 0 || qw > 254 || (qw & 1)) continue;
            acc = fmaf(ch * c_g[3 - jw], row[qw >> 1], acc);
        }
    }
    out[idx] = acc;
}

// ---------------------------------------------------------------------------
// Host side
// ---------------------------------------------------------------------------
static inline char* carve(char*& cur, size_t bytes)
{
    char* r = cur;
    cur += (bytes + 255) & ~(size_t)255;
    return r;
}

extern "C" void kernel_launch(void* const* d_in, const int* in_sizes, int n_in,
                              void* d_out, int out_size, void* d_ws, size_t ws_size,
                              hipStream_t stream)
{
    const float* x        = (const float*)d_in[0];   // (8,64,256,256)
    const float* conv1_w  = (const float*)d_in[1];   // (128,64,3,3)
    const float* conv1_b  = (const float*)d_in[2];
    const float* bn1_g    = (const float*)d_in[3];
    const float* bn1_b    = (const float*)d_in[4];
    const float* conv2_w  = (const float*)d_in[5];   // (128,128,3,3)
    const float* conv2_b  = (const float*)d_in[6];
    const float* bn2_g    = (const float*)d_in[7];
    const float* bn2_b    = (const float*)d_in[8];
    const float* deconv_w = (const float*)d_in[9];   // (128,64,2,2)
    const float* deconv_b = (const float*)d_in[10];  // (64)
    float* out = (float*)d_out;
    (void)in_sizes; (void)n_in; (void)out_size; (void)ws_size;

    const int N = 8, C0 = 64, CM = 128, H = 64, W = 64, HW = H * W;

    // Workspace carving
    char* cur = (char*)d_ws;
    _Float16* P16  = (_Float16*)carve(cur, (size_t)N * C0 * HW * 2);        // f16 NHWC
    float*    C1   = (float*)   carve(cur, (size_t)N * CM * HW * 4);        // f32 NCHW
    _Float16* A1   = (_Float16*)carve(cur, (size_t)N * CM * HW * 2);        // f16 NHWC
    float*    C2   = (float*)   carve(cur, (size_t)N * CM * HW * 4);
    _Float16* A2   = (_Float16*)carve(cur, (size_t)N * CM * HW * 2);
    float*    Yd   = (float*)   carve(cur, (size_t)N * C0 * 128 * 128 * 4); // f32 NCHW
    _Float16* PW1  = (_Float16*)carve(cur, (size_t)CM * C0 * 9 * 2);
    _Float16* PW2  = (_Float16*)carve(cur, (size_t)CM * CM * 9 * 2);
    _Float16* PWD  = (_Float16*)carve(cur, (size_t)256 * 128 * 2);
    float*    mean = (float*)   carve(cur, CM * 4);
    float*    var  = (float*)   carve(cur, CM * 4);

    // 0. Weight packing (tiny)
    pack_conv_w_kernel<64, 128><<<(128 * 64 * 9 + 255) / 256, 256, 0, stream>>>(conv1_w, PW1);
    pack_conv_w_kernel<128, 128><<<(128 * 128 * 9 + 255) / 256, 256, 0, stream>>>(conv2_w, PW2);
    pack_deconv_w_kernel<<<(256 * 128 + 255) / 256, 256, 0, stream>>>(deconv_w, PWD);

    // 1. DWT-LL + maxpool -> f16 NHWC
    {
        long total = (long)N * C0 * HW;
        dwt_pool_kernel<<<(int)((total + 255) / 256), 256, 0, stream>>>(x, P16, C0, total);
    }
    // 2. conv1 (64->128) via WMMA
    {
        dim3 grid((N * HW) / 16, CM / 16);
        conv3_wmma_kernel<64, 128><<<grid, 32, 0, stream>>>(P16, PW1, conv1_b, C1, H, W);
    }
    // 3. BN1 stats + normalize + ReLU -> f16 NHWC
    {
        bn_stats_kernel<<<CM, 256, 0, stream>>>(C1, mean, var, N, CM, HW);
        long total = (long)N * CM * HW;
        bn_relu_kernel<<<(int)((total + 255) / 256), 256, 0, stream>>>(
            C1, A1, mean, var, bn1_g, bn1_b, CM, H, W, total);
    }
    // 4. conv2 (128->128) via WMMA
    {
        dim3 grid((N * HW) / 16, CM / 16);
        conv3_wmma_kernel<128, 128><<<grid, 32, 0, stream>>>(A1, PW2, conv2_b, C2, H, W);
    }
    // 5. BN2 stats + normalize + ReLU -> f16 NHWC
    {
        bn_stats_kernel<<<CM, 256, 0, stream>>>(C2, mean, var, N, CM, HW);
        long total = (long)N * CM * HW;
        bn_relu_kernel<<<(int)((total + 255) / 256), 256, 0, stream>>>(
            C2, A2, mean, var, bn2_g, bn2_b, CM, H, W, total);
    }
    // 6. deconv 2x upsample (128->64) via WMMA GEMM
    {
        dim3 grid((N * HW) / 16, 256 / 16);
        deconv2_wmma_kernel<<<grid, 32, 0, stream>>>(A2, PWD, deconv_b, Yd);
    }
    // 7. inverse DWT-LL -> final output (8,64,254,254)
    {
        long total = (long)N * C0 * 254 * 254;
        idwt2_kernel<<<(int)((total + 255) / 256), 256, 0, stream>>>(Yd, out, total);
    }
}
